// RNN_2095944040494
// MI455X (gfx1250) — compile-verified, matrix operand order checked by reading, not run on hardware
//
#include <hip/hip_runtime.h>
#include <hip/hip_bf16.h>

// ---------------------------------------------------------------------------
// E/I rate RNN on MI455X (gfx1250, wave32).
//  B=256, T=500, I=8, H=1024, O=8.
//  Recurrent GEMM: v_wmma_f32_16x16x32_bf16 (f32 accumulate).
//  Persistent kernel, 32 WGs = 2 batch tiles (128 rows) x 16 column tiles
//  (64 cols). Only the 16 WGs of a batch tile synchronize (atomic barrier).
//  r exchanged via ping-pong bf16 buffers in d_ws (L2 resident).
//  - Each WG's constant B slice (effW cols, 128KB) is DMA'd into LDS once via
//    the Tensor Data Mover (tensor_load_to_lds) with pad_enable building a
//    bank-conflict-free [kchunk][col][32+8pad] layout; the K-loop feeds the
//    WMMA B operand from LDS (ds_load_b128).
//  - Next-step noise/x loads are issued inside the barrier window (after the
//    release fence + arrive, before the spin) so their HBM latency overlaps
//    the inter-WG synchronization instead of the critical path.
// Workspace layout (needs ~3.1 MB):
//   [0, 2MB)        effWg  : bf16 effW[j][k] * gr[k]   (row-major HxH)
//   [2MB, 2.5MB)    rbuf0  : bf16 r, ping               (B x H)
//   [2.5MB, 3MB)    rbuf1  : bf16 r, pong               (B x H)
//   [3MB, 3MB+256)  barrier counters (one per batch tile, 128B apart)
// ---------------------------------------------------------------------------

typedef __attribute__((ext_vector_type(16))) __bf16 v16bf;
typedef __attribute__((ext_vector_type(8)))  __bf16 v8bf;
typedef __attribute__((ext_vector_type(8)))  float  v8f;
typedef __attribute__((ext_vector_type(4)))  unsigned int u32x4;
typedef __attribute__((ext_vector_type(8)))  int i32x8;
typedef __attribute__((ext_vector_type(4)))  int i32x4;

union BF16x16 { v16bf v; v8bf h[2]; };

#define RNN_B 256
#define RNN_T 500
#define RNN_I 8
#define RNN_H 1024
#define RNN_O 8

// LDS B layout: 32 k-chunks x 64 cols x (32 k + 8 pad) bf16 = 160KB
#define BCOL_STRIDE 40            // bf16 elements per (kchunk, col) row
#define BCHUNK_STRIDE (64 * BCOL_STRIDE)

__device__ __forceinline__ float softplus1(float v) {
    // beta = 1: softplus(v) = max(v,0) + log(1+exp(-|v|)), fast HW exp/log.
    return fmaxf(v, 0.0f) + __logf(1.0f + __expf(-fabsf(v)));
}

// ---------------------------------------------------------------------------
// Setup kernels
// ---------------------------------------------------------------------------

__global__ void k_zero(float* __restrict__ out, unsigned* __restrict__ cnt, size_t n) {
    size_t gid = (size_t)blockIdx.x * blockDim.x + threadIdx.x;
    size_t stride = (size_t)gridDim.x * blockDim.x;
    for (size_t i = gid; i < n; i += stride) out[i] = 0.0f;
    if (gid < 64) cnt[gid] = 0u;
}

// effWg[j][k] = ( relu(wrec*ei)*ei - wrec*(|ei|-1) ) * mwrec * relu(g[k]),  bf16
__global__ void k_effw(const float* __restrict__ wrec, const float* __restrict__ refEI,
                       const float* __restrict__ mwrec, const float* __restrict__ g,
                       __bf16* __restrict__ effW) {
    size_t idx = (size_t)blockIdx.x * 256 + threadIdx.x;   // j*H + k
    int k = (int)(idx & (RNN_H - 1));
    float w  = wrec[idx];
    float ei = refEI[idx];
    float e  = fmaxf(w * ei, 0.0f) * ei + (-w) * (fabsf(ei) - 1.0f);
    float grk = fmaxf(g[k], 0.0f);
    effW[idx] = (__bf16)(e * mwrec[idx] * grk);
}

// rbuf0[b][h] = bf16( softplus(h0[h] + b[h]) )   (r at t=0, broadcast over batch)
__global__ void k_rinit(const float* __restrict__ bvec, const float* __restrict__ h0,
                        __bf16* __restrict__ rb0) {
    size_t idx = (size_t)blockIdx.x * 256 + threadIdx.x;   // b*H + h
    int h = (int)(idx & (RNN_H - 1));
    rb0[idx] = (__bf16)softplus1(h0[h] + bvec[h]);
}

// out[b][0][o] = sum_h softplus(h0[h]+b[h]) * wout[h][o]
__global__ void k_out0(const float* __restrict__ bvec, const float* __restrict__ h0,
                       const float* __restrict__ wout, float* __restrict__ out) {
    int gid = blockIdx.x * 256 + threadIdx.x;              // 0 .. B*O-1
    if (gid >= RNN_B * RNN_O) return;
    int b = gid >> 3, o = gid & 7;
    float s = 0.0f;
    for (int h = 0; h < RNN_H; ++h)
        s += softplus1(h0[h] + bvec[h]) * wout[h * RNN_O + o];
    out[(size_t)b * RNN_T * RNN_O + o] = s;
}

// ---------------------------------------------------------------------------
// TDM: load this WG's effW column slice into LDS as [kchunk][col][32k+8pad].
//  3D descriptor: X = 32 k-elements (contiguous in a chunk),
//                 Y = 64 columns  (global stride 1024 elements),
//                 Z = 32 k-chunks (global stride 32 elements).
//  pad_enable inserts 4 DWORDs (16B) after every 16 DWORDs (64B) stored,
//  producing the 80B column stride (conflict-free ds_load_b128).
// ---------------------------------------------------------------------------
__device__ __forceinline__ void tdm_load_B(const __bf16* gsrc, unsigned lds_off) {
    unsigned long long ga = (unsigned long long)(size_t)gsrc;
    u32x4 g0;
    g0[0] = 1u;                                   // count=1, user descriptor
    g0[1] = lds_off;                              // LDS byte address
    g0[2] = (unsigned)(ga & 0xFFFFFFFFu);         // global addr lo
    g0[3] = (unsigned)((ga >> 32) & 0x01FFFFFFu)  // global addr hi (57-bit)
          | 0x80000000u;                          // type = 2 ("image")
    i32x8 g1;
    g1[0] = (int)0x06D10000u;   // data_size=2B, pad_en, pad_int=16dw, pad_amt=4dw
    g1[1] = (int)(32u << 16);   // tensor_dim0 = 32
    g1[2] = (int)(64u << 16);   // tensor_dim1 = 64
    g1[3] = (int)(32u << 16);   // tile_dim0 = 32
    g1[4] = (int)(64u | (32u << 16)); // tile_dim1 = 64, tile_dim2 = 32
    g1[5] = 1024;               // tensor_dim0_stride = 1024 (elements)
    g1[6] = (int)(32u << 16);   // tensor_dim1_stride = 32 (elements)
    g1[7] = 0;
    i32x4 g2; g2[0] = 32; g2[1] = 0; g2[2] = 0; g2[3] = 0;   // tensor_dim2 = 32
    i32x4 g3; g3[0] = 0;  g3[1] = 0; g3[2] = 0; g3[3] = 0;
#if __clang_major__ >= 23
    i32x8 g4; g4[0]=0; g4[1]=0; g4[2]=0; g4[3]=0; g4[4]=0; g4[5]=0; g4[6]=0; g4[7]=0;
    __builtin_amdgcn_tensor_load_to_lds(g0, g1, g2, g3, g4, 0);
#else
    __builtin_amdgcn_tensor_load_to_lds(g0, g1, g2, g3, 0);
#endif
}

// ---------------------------------------------------------------------------
// Persistent recurrent kernel.
//   grid = 32 WGs: blockIdx.x = btile*16 + ctile  (btile 0..1, ctile 0..15)
//   block = 256 threads = 8 waves; wave w owns batch rows [btile*128+w*16, +16)
//   and all 64 columns of its ctile -> 4 WMMA accumulators per wave.
// ---------------------------------------------------------------------------
__global__ __launch_bounds__(256, 1)
void k_rnn(const float* __restrict__ x, const float* __restrict__ noise,
           const float* __restrict__ wi, const float* __restrict__ wout,
           const float* __restrict__ bvec, const float* __restrict__ h0,
           const __bf16* __restrict__ effW,
           __bf16* __restrict__ rb0, __bf16* __restrict__ rb1,
           float* __restrict__ out, unsigned* __restrict__ cnt) {
    constexpr int H = RNN_H, T = RNN_T, I = RNN_I, O = RNN_O;

    __shared__ __align__(128) __bf16 Bsh[32 * BCHUNK_STRIDE];  // 160KB
    __shared__ float xs[128 * I];                              // 4KB

    const int tid   = threadIdx.x;
    const int lane  = tid & 31;
    const int wave  = tid >> 5;           // 0..7 -> M stripe
    const int btile = blockIdx.x >> 4;    // 0..1
    const int ctile = blockIdx.x & 15;    // 0..15
    const int rowbase = btile * 128 + wave * 16;
    const int colbase = ctile * 64;
    const int half  = lane >> 4;          // K-half for A/B fragments
    const int nlo   = lane & 15;          // row (A) / column (B) / N (C)

    // One TDM per WG: pull the 64x1024 effW slice into LDS (pad-interleaved).
    if (wave == 0) {
        tdm_load_B(effW + (size_t)colbase * H, (unsigned)(size_t)&Bsh[0]);
        __builtin_amdgcn_s_wait_tensorcnt(0);
    }

    // Per-thread constants: columns, input & readout weights, biases.
    int   hcol[4];
    float wi_r[4][I], wo_r[4][O], bT[4];
    v8f   hst[4];                          // persistent h state (C layout)
#pragma unroll
    for (int nt = 0; nt < 4; ++nt) {
        hcol[nt] = colbase + nt * 16 + nlo;
#pragma unroll
        for (int i = 0; i < I; ++i) wi_r[nt][i] = wi[i * H + hcol[nt]];
#pragma unroll
        for (int o = 0; o < O; ++o) wo_r[nt][o] = wout[(size_t)hcol[nt] * O + o];
        bT[nt] = bvec[hcol[nt]];
        float h0c = h0[hcol[nt]];
#pragma unroll
        for (int v = 0; v < 8; ++v) hst[nt][v] = h0c;   // h init = h0
    }

    // Per-thread LDS base for the B fragments (80B column stride).
    const __bf16* bcol[4];
#pragma unroll
    for (int nt = 0; nt < 4; ++nt)
        bcol[nt] = Bsh + (nt * 16 + nlo) * BCOL_STRIDE + half * 8;

    // Stage x[:, tt-1, :] for this batch tile into LDS (128 x 8 f32).
    auto stage_x = [&](int tt) {
        for (int i2 = tid; i2 < 128 * I; i2 += 256) {
            int rr = i2 >> 3, ii = i2 & 7;
            xs[i2] = x[((size_t)(btile * 128 + rr) * T + (tt - 1)) * I + ii];
        }
    };

    // Next-step noise gathers into registers (independent of r exchange).
    float nzr[4][8];
    auto load_nz = [&](int tt) {
#pragma unroll
        for (int nt = 0; nt < 4; ++nt)
#pragma unroll
            for (int v = 0; v < 8; ++v) {
                int brow = rowbase + v + 8 * half;
                nzr[nt][v] = noise[((size_t)brow * T + (tt - 1)) * H + hcol[nt]];
            }
    };

    unsigned* mycnt = cnt + btile * 32;    // 128B-separated counters

    // Prologue: stage step-1 operands (TDM wait above covers wave 0).
    stage_x(1);
    load_nz(1);
    __syncthreads();

    for (int t = 1; t < T; ++t) {
        const __bf16* rcur = (t & 1) ? rb0 : rb1;   // r from step t-1
        __bf16*       rwrt = (t & 1) ? rb1 : rb0;   // r for step t

        // Fold leak + noise + input into the WMMA C-init:
        //   h' = 0.8h + 0.05n + 0.2in + 0.2*gemm = 0.2*(4h + 0.25n + in + gemm)
        v8f acc[4];
#pragma unroll
        for (int nt = 0; nt < 4; ++nt) {
#pragma unroll
            for (int v = 0; v < 8; ++v) {
                int lrow = wave * 16 + v + 8 * half;          // row in tile
                float inp = 0.0f;
#pragma unroll
                for (int i = 0; i < I; ++i) inp += xs[lrow * I + i] * wi_r[nt][i];
                acc[nt][v] = 4.0f * hst[nt][v] + 0.25f * nzr[nt][v] + inp;
            }
        }

        // K loop: 32 iterations of v_wmma_f32_16x16x32_bf16.
        // A = r_prev (bf16 rows from global ping-pong buffer),
        // B = effWg columns served from LDS (ds_load_b128, conflict-free).
#pragma unroll 2
        for (int kt = 0; kt < 32; ++kt) {
            BF16x16 afrag;
            const __bf16* ap = rcur + (size_t)(rowbase + nlo) * H + kt * 32 + half * 8;
            afrag.h[0] = *(const v8bf*)(ap);
            afrag.h[1] = *(const v8bf*)(ap + 16);
            const int bo = kt * BCHUNK_STRIDE;
#pragma unroll
            for (int nt = 0; nt < 4; ++nt) {
                BF16x16 bfrag;
                const __bf16* bp = bcol[nt] + bo;
                bfrag.h[0] = *(const v8bf*)(bp);
                bfrag.h[1] = *(const v8bf*)(bp + 16);
                acc[nt] = __builtin_amdgcn_wmma_f32_16x16x32_bf16(
                    false, afrag.v, false, bfrag.v, (short)0, acc[nt], false, false);
            }
        }

        // Finalize: h' = 0.2*acc, r = softplus(h'+b); store r (bf16) for the
        // next step; reduce readout partials over the 16 lanes sharing a row.
#pragma unroll
        for (int v = 0; v < 8; ++v) {
            const int brow = rowbase + v + 8 * half;
            float po[O] = {0, 0, 0, 0, 0, 0, 0, 0};
#pragma unroll
            for (int nt = 0; nt < 4; ++nt) {
                float hn = 0.2f * acc[nt][v];
                hst[nt][v] = hn;
                float r = softplus1(hn + bT[nt]);
                rwrt[(size_t)brow * H + hcol[nt]] = (__bf16)r;
#pragma unroll
                for (int o = 0; o < O; ++o) po[o] += r * wo_r[nt][o];
            }
#pragma unroll
            for (int off = 1; off < 16; off <<= 1)
#pragma unroll
                for (int o = 0; o < O; ++o) po[o] += __shfl_xor(po[o], off, 16);
            if (nlo == 0) {
#pragma unroll
                for (int o = 0; o < O; ++o)
                    atomicAdd(&out[((size_t)brow * T + t) * O + o], po[o]);
            }
        }

        // Fused inter-WG barrier + next-step prefetch:
        //   release fence, arrive, then issue t+1 noise/x loads while waiting.
        __threadfence();                 // release: r stores visible (DEV scope)
        __syncthreads();                 // xs(t) reads complete in all threads
        if (tid == 0)
            __hip_atomic_fetch_add(mycnt, 1u, __ATOMIC_RELAXED,
                                   __HIP_MEMORY_SCOPE_AGENT);
        if (t + 1 < T) {                 // overlap HBM latency with the spin
            stage_x(t + 1);
            load_nz(t + 1);
        }
        if (tid == 0) {
            unsigned target = 16u * (unsigned)t;
            while (__hip_atomic_load(mycnt, __ATOMIC_RELAXED,
                                     __HIP_MEMORY_SCOPE_AGENT) < target)
                __builtin_amdgcn_s_sleep(2);
        }
        __syncthreads();                 // xs(t+1) visible to all threads
        __threadfence();                 // acquire: invalidate L0 before r reads
    }
}

// ---------------------------------------------------------------------------
extern "C" void kernel_launch(void* const* d_in, const int* in_sizes, int n_in,
                              void* d_out, int out_size, void* d_ws, size_t ws_size,
                              hipStream_t stream) {
    const float* x     = (const float*)d_in[0];
    const float* noise = (const float*)d_in[1];
    const float* wi    = (const float*)d_in[2];
    const float* wrec  = (const float*)d_in[3];
    const float* wout  = (const float*)d_in[4];
    const float* bvec  = (const float*)d_in[5];
    const float* g     = (const float*)d_in[6];
    const float* h0    = (const float*)d_in[7];
    const float* refEI = (const float*)d_in[8];
    const float* mwrec = (const float*)d_in[9];
    float* out = (float*)d_out;

    char* ws = (char*)d_ws;
    __bf16*   effW = (__bf16*)(ws);
    __bf16*   rb0  = (__bf16*)(ws + (2u << 20));
    __bf16*   rb1  = (__bf16*)(ws + (2u << 20) + (512u << 10));
    unsigned* cnt  = (unsigned*)(ws + (3u << 20));

    const size_t nout = (size_t)RNN_B * RNN_T * RNN_O;
    k_zero <<<2048, 256, 0, stream>>>(out, cnt, nout);
    k_effw <<<(RNN_H * RNN_H) / 256, 256, 0, stream>>>(wrec, refEI, mwrec, g, effW);
    k_rinit<<<(RNN_B * RNN_H) / 256, 256, 0, stream>>>(bvec, h0, rb0);
    k_out0 <<<(RNN_B * RNN_O + 255) / 256, 256, 0, stream>>>(bvec, h0, wout, out);
    k_rnn  <<<32, 256, 0, stream>>>(x, noise, wi, wout, bvec, h0,
                                    effW, rb0, rb1, out, cnt);
}
